// BiLSTM_stacked_18408229831106
// MI455X (gfx1250) — compile-verified, matrix operand order checked by reading, not run on hardware
//
#include <hip/hip_runtime.h>
#include <hip/hip_bf16.h>

// ---------------------------------------------------------------------------
// Stacked BiLSTM for MI455X (gfx1250, wave32, WMMA).
//
//  1) lengths[b] from x[:,:,0]
//  2) convert x (and each W*_ih) to bf16
//  3) x_proj = A @ W^T + b via v_wmma_f32_16x16x32_bf16, 4x4 register-blocked
//     (64x64 output per wave -> 4x less L2 fetch traffic than 1 tile/wave)
//  4) recurrence: persistent 16-workgroup cluster kernel, one WGP per 32
//     hidden units; Whh slice resident in LDS as bf16 (128KB, uses CDNA5's
//     320KB LDS); per-step sync = CDNA5 cluster barrier + agent fences;
//     per-step gate GEMM (64x512 @ 512x128) via WMMA from LDS with A-frag
//     reuse across the wave's two n-tiles.
//  5) head: out = (h2f+h2b) @ W3^T + b3
// ---------------------------------------------------------------------------

#define Bb   64
#define Tt   512
#define Dd   768
#define Hh   512
#define G4H  2048   // 4*H
#define OUTN 256

typedef __bf16 bf16x16 __attribute__((ext_vector_type(16)));
typedef __bf16 bf16x8  __attribute__((ext_vector_type(8)));
typedef float  f32x8   __attribute__((ext_vector_type(8)));

__device__ __forceinline__ float sigmoidf_(float x) {
    return 1.0f / (1.0f + __expf(-x));
}

// ---------------------------------------------------------------------------
__global__ void lengths_kernel(const float* __restrict__ x, int* __restrict__ len) {
    int b = blockIdx.x * blockDim.x + threadIdx.x;
    if (b >= Bb) return;
    int c = 0;
    for (int t = 0; t < Tt; ++t)
        c += (x[((size_t)b * Tt + t) * Dd] != 0.0f) ? 1 : 0;
    len[b] = c;
}

__global__ void f32_to_bf16_kernel(const float* __restrict__ in, __bf16* __restrict__ out, int n) {
    int i = blockIdx.x * blockDim.x + threadIdx.x;
    if (i < n) out[i] = (__bf16)in[i];
}

__global__ void zero_f32_kernel(float* __restrict__ p, int n) {
    int i = blockIdx.x * blockDim.x + threadIdx.x;
    if (i < n) p[i] = 0.0f;
}

// ---------------------------------------------------------------------------
// C[M,N] = A[M,K](bf16) @ W[N,K]^T(bf16) + bias[N], f32 accumulate.
// 4x4 register blocking: each wave computes a 64x64 output block
// (16 accumulators). Per 32-wide K-chunk: 8 fragment loads feed 16 WMMAs.
// Fragment addressing follows the CDNA5 ISA 16-bit A-matrix 16x32 layout
// (lanes 0-15: K 0..7 & 16..23, lanes 16-31: K 8..15 & 24..31).
// M % 64 == 0 and N % 64 == 0 required (holds: M=32768, N=2048).
// ---------------------------------------------------------------------------
__global__ __launch_bounds__(256)
void gemm_bf16_wmma_kernel(const __bf16* __restrict__ A, const __bf16* __restrict__ W,
                           const float* __restrict__ bias, float* __restrict__ C,
                           int M, int N, int K)
{
    const int lane = threadIdx.x & 31;
    const int wv   = threadIdx.x >> 5;
    const int nwt  = N >> 6;                      // 64-wide wave tiles in N
    const int wt   = blockIdx.x * 8 + wv;
    const int m0   = (wt / nwt) * 64;
    const int n0   = (wt % nwt) * 64;
    if (m0 >= M) return;

    const int rsel = lane & 15;
    const int kb   = (lane < 16) ? 0 : 8;

    const __bf16* apB = A + (size_t)(m0 + rsel) * K + kb;
    const __bf16* bpB = W + (size_t)(n0 + rsel) * K + kb;

    const f32x8 zv = {};
    f32x8 acc[4][4];
#pragma unroll
    for (int i = 0; i < 4; ++i)
#pragma unroll
        for (int j = 0; j < 4; ++j) acc[i][j] = zv;

    for (int k0 = 0; k0 < K; k0 += 32) {
        union { bf16x16 v; bf16x8 h[2]; } Af[4], Bf[4];
#pragma unroll
        for (int i = 0; i < 4; ++i) {
            const __bf16* ap = apB + (size_t)(i * 16) * K + k0;
            Af[i].h[0] = *(const bf16x8*)(ap);
            Af[i].h[1] = *(const bf16x8*)(ap + 16);
            const __bf16* bp = bpB + (size_t)(i * 16) * K + k0;
            Bf[i].h[0] = *(const bf16x8*)(bp);
            Bf[i].h[1] = *(const bf16x8*)(bp + 16);
        }
        __builtin_prefetch(apB + k0 + 256, 0, 1);   // global_prefetch ahead
        __builtin_prefetch(bpB + k0 + 256, 0, 1);
#pragma unroll
        for (int i = 0; i < 4; ++i)
#pragma unroll
            for (int j = 0; j < 4; ++j)
                acc[i][j] = __builtin_amdgcn_wmma_f32_16x16x32_bf16(
                                false, Af[i].v, false, Bf[j].v, (short)0,
                                acc[i][j], false, false);
    }

    // C/D layout: VGPR r, lanes 0-15 -> M=r, lanes 16-31 -> M=r+8; N = lane&15
    const int gmB = m0 + ((lane < 16) ? 0 : 8);
    const int gnB = n0 + (lane & 15);
#pragma unroll
    for (int j = 0; j < 4; ++j) {
        const int gn = gnB + j * 16;
        const float bb = bias[gn];
#pragma unroll
        for (int i = 0; i < 4; ++i) {
            const int gm = gmB + i * 16;
#pragma unroll
            for (int r = 0; r < 8; ++r)
                C[(size_t)(gm + r) * N + gn] = acc[i][j][r] + bb;
        }
    }
}

// ---------------------------------------------------------------------------
// Persistent recurrent kernel: grid = 16 workgroups forming one cluster,
// 512 threads (16 waves) each. Block k owns hidden units [k*32, k*32+32)
// i.e. gate rows {q*512 + k*32 + r}. Whh slice stays in LDS (bf16) for all
// T steps. Per step:
//   - reload full h (64x512) from global into LDS (bf16)
//   - gates(64x128) = h @ Whh_slice^T via WMMA; each wave owns 2 n-adjacent
//     tiles sharing one A fragment per K-chunk
//   - epilogue: gates += xproj[t] (bias folded in), activations, c/h update,
//     masked by s < len[b]; write h slice to global; optional out1 write
//   - agent release fence -> cluster barrier -> agent acquire fence
//
// dir: 0 = forward (t = s), 1 = backward (t = len[b]-1-s)
// out_mode: 0 = none, 1 = set (also zero-fill masked-out), 2 = accumulate
// ---------------------------------------------------------------------------
#define WS_STRIDE 520                     // 512 + 8 bf16 pad (bank-conflict free)
#define HS_STRIDE 520
#define GS_STRIDE 132                     // 128 + 4 f32 pad
#define WS_BYTES  (128 * WS_STRIDE * 2)   // 133120
#define HS_BYTES  (64  * HS_STRIDE * 2)   //  66560
#define GS_BYTES  (64  * GS_STRIDE * 4)   //  33792
#define CS_BYTES  (64  * 32 * 4)          //   8192
#define REC_SMEM  (WS_BYTES + HS_BYTES + GS_BYTES + CS_BYTES)  // 241664 B

__global__ __launch_bounds__(512)
void lstm_rec_kernel(const float* __restrict__ xproj,  // [B,T,4H] (bias included)
                     const float* __restrict__ Whh,    // [4H,H] f32
                     const int*   __restrict__ len,
                     float*       __restrict__ hglob,  // [B,H] f32, pre-zeroed
                     float*       __restrict__ outp,   // [B,T,H] or nullptr
                     int dir, int out_mode)
{
    extern __shared__ char smem_raw[];
    __bf16* ws = (__bf16*)(smem_raw);                       // [128][WS_STRIDE]
    __bf16* hs = (__bf16*)(smem_raw + WS_BYTES);            // [64][HS_STRIDE]
    float*  gs = (float*) (smem_raw + WS_BYTES + HS_BYTES); // [64][GS_STRIDE]
    float*  cs = (float*) (smem_raw + WS_BYTES + HS_BYTES + GS_BYTES); // [64][32]

    const int tid  = threadIdx.x;
    const int lane = tid & 31;
    const int wv   = tid >> 5;
    const int blk  = blockIdx.x;          // 0..15 : hidden-unit chunk

    // Load this block's Whh slice into LDS as bf16:
    // local row nl = q*32 + r  <->  global gate row q*512 + blk*32 + r
    for (int idx = tid; idx < 128 * Hh; idx += 512) {
        const int nl = idx >> 9;           // 0..127
        const int k  = idx & 511;
        const int q  = nl >> 5, r = nl & 31;
        const int grow = q * Hh + blk * 32 + r;   // q*512 + ...
        ws[nl * WS_STRIDE + k] = (__bf16)Whh[(size_t)grow * Hh + k];
    }
    for (int idx = tid; idx < 64 * 32; idx += 512) cs[idx] = 0.0f;
    __syncthreads();

    // Each thread's 4 epilogue slots: b = wv + 16*u, r = lane
    int Lb[4];
#pragma unroll
    for (int u = 0; u < 4; ++u) Lb[u] = len[wv + 16 * u];

    // Wave's two tiles (2*wv, 2*wv+1) always share mt -> share the A fragment.
    const int tile0 = wv * 2;
    const int mt    = tile0 >> 3;          // 0..3
    const int nt0   = tile0 & 7;           // even: 0,2,4,6
    const int arow  = mt  * 16 + (lane & 15);
    const int brow0 = nt0 * 16 + (lane & 15);
    const int kb    = (lane < 16) ? 0 : 8;

    const f32x8 zv = {};

    for (int s = 0; s < Tt; ++s) {
        // ---- broadcast h into LDS (bf16) -------------------------------
        for (int idx = tid; idx < Bb * Hh; idx += 512) {
            const int b = idx >> 9, k = idx & 511;
            hs[b * HS_STRIDE + k] = (__bf16)hglob[idx];
        }
        __syncthreads();

        // ---- gates(64x128) = h @ Whh_slice^T ---------------------------
        {
            const __bf16* ap  = hs + arow  * HS_STRIDE + kb;
            const __bf16* bp0 = ws + brow0 * WS_STRIDE + kb;
            const __bf16* bp1 = bp0 + 16 * WS_STRIDE;
            f32x8 acc0 = zv, acc1 = zv;
#pragma unroll
            for (int kk = 0; kk < Hh / 32; ++kk) {
                union { bf16x16 v; bf16x8 h[2]; } Af, B0, B1;
                Af.h[0] = *(const bf16x8*)(ap);
                Af.h[1] = *(const bf16x8*)(ap + 16);
                B0.h[0] = *(const bf16x8*)(bp0);
                B0.h[1] = *(const bf16x8*)(bp0 + 16);
                B1.h[0] = *(const bf16x8*)(bp1);
                B1.h[1] = *(const bf16x8*)(bp1 + 16);
                acc0 = __builtin_amdgcn_wmma_f32_16x16x32_bf16(
                           false, Af.v, false, B0.v, (short)0, acc0, false, false);
                acc1 = __builtin_amdgcn_wmma_f32_16x16x32_bf16(
                           false, Af.v, false, B1.v, (short)0, acc1, false, false);
                ap  += 32;
                bp0 += 32;
                bp1 += 32;
            }
            const int gm = mt * 16 + ((lane < 16) ? 0 : 8);
            const int gn0 = nt0 * 16 + (lane & 15);
#pragma unroll
            for (int r = 0; r < 8; ++r) {
                gs[(gm + r) * GS_STRIDE + gn0]      = acc0[r];
                gs[(gm + r) * GS_STRIDE + gn0 + 16] = acc1[r];
            }
        }
        __syncthreads();

        // ---- pointwise LSTM cell for our 32 hidden units ---------------
#pragma unroll
        for (int u = 0; u < 4; ++u) {
            const int b = wv + 16 * u;
            const int r = lane;
            const int L = Lb[u];
            const int col = blk * 32 + r;           // hidden index
            const bool act = (s < L);
            if (act) {
                const int t = (dir == 0) ? s : (L - 1 - s);
                const float* xp = xproj + ((size_t)b * Tt + t) * G4H;
                const float gi = gs[b * GS_STRIDE +  0 + r] + xp[         col];
                const float gf = gs[b * GS_STRIDE + 32 + r] + xp[ Hh    + col];
                const float gg = gs[b * GS_STRIDE + 64 + r] + xp[ 2*Hh  + col];
                const float go = gs[b * GS_STRIDE + 96 + r] + xp[ 3*Hh  + col];
                const float i_ = sigmoidf_(gi);
                const float f_ = sigmoidf_(gf);
                const float g_ = tanhf(gg);
                const float o_ = sigmoidf_(go);
                const float cn = f_ * cs[b * 32 + r] + i_ * g_;
                const float hn = o_ * tanhf(cn);
                cs[b * 32 + r] = cn;
                hglob[(size_t)b * Hh + col] = hn;
                if (out_mode == 1)
                    outp[((size_t)b * Tt + t) * Hh + col] = hn;
                else if (out_mode == 2)
                    outp[((size_t)b * Tt + t) * Hh + col] += hn;
            } else if (out_mode == 1) {
                outp[((size_t)b * Tt + s) * Hh + col] = 0.0f;
            }
        }

        // ---- cross-WGP sync: release -> cluster barrier -> acquire -----
        __builtin_amdgcn_fence(__ATOMIC_RELEASE, "agent");
        __syncthreads();
        if (wv == 0)
            __builtin_amdgcn_s_cluster_barrier();   // s_barrier_signal/wait -3
        __syncthreads();
        __builtin_amdgcn_fence(__ATOMIC_ACQUIRE, "agent");
    }
}

// ---------------------------------------------------------------------------
__global__ void head_kernel(const float* __restrict__ h2f, const float* __restrict__ h2b,
                            const float* __restrict__ W3, const float* __restrict__ b3,
                            float* __restrict__ out)
{
    const int o = blockIdx.x * blockDim.x + threadIdx.x;
    if (o >= Bb * OUTN) return;
    const int b = o >> 8;
    const int n = o & 255;
    float acc = b3[n];
    const float* hf = h2f + (size_t)b * Hh;
    const float* hb = h2b + (size_t)b * Hh;
    const float* w  = W3  + (size_t)n * Hh;
    for (int k = 0; k < Hh; ++k)
        acc += (hf[k] + hb[k]) * w[k];
    out[o] = acc;
}

// ---------------------------------------------------------------------------
extern "C" void kernel_launch(void* const* d_in, const int* in_sizes, int n_in,
                              void* d_out, int out_size, void* d_ws, size_t ws_size,
                              hipStream_t stream)
{
    (void)in_sizes; (void)n_in; (void)out_size; (void)ws_size;
    const float* x      = (const float*)d_in[0];
    const float* W1f_ih = (const float*)d_in[1];
    const float* W1f_hh = (const float*)d_in[2];
    const float* b1f    = (const float*)d_in[3];
    const float* W1b_ih = (const float*)d_in[4];
    const float* W1b_hh = (const float*)d_in[5];
    const float* b1b    = (const float*)d_in[6];
    const float* W2f_ih = (const float*)d_in[7];
    const float* W2f_hh = (const float*)d_in[8];
    const float* b2f    = (const float*)d_in[9];
    const float* W2b_ih = (const float*)d_in[10];
    const float* W2b_hh = (const float*)d_in[11];
    const float* b2b    = (const float*)d_in[12];
    const float* W3     = (const float*)d_in[13];
    const float* b3     = (const float*)d_in[14];
    float* out = (float*)d_out;

    // ---- workspace carve-up (all offsets 256B aligned) ---------------------
    char* wsp = (char*)d_ws;
    size_t cur = 0;
    auto carve = [&](size_t bytes) -> void* {
        void* p = wsp + cur;
        cur += (bytes + 255) & ~(size_t)255;
        return p;
    };
    int*    d_len   = (int*)   carve(Bb * sizeof(int));
    __bf16* xbf     = (__bf16*)carve((size_t)Bb * Tt * Dd * 2);      //  48 MB
    float*  out1    = (float*) carve((size_t)Bb * Tt * Hh * 4);      //  64 MB
    __bf16* out1bf  = (__bf16*)carve((size_t)Bb * Tt * Hh * 2);      //  32 MB
    float*  xproj   = (float*) carve((size_t)Bb * Tt * G4H * 4);     // 256 MB
    __bf16* wbf     = (__bf16*)carve((size_t)G4H * Dd * 2);          //   3 MB
    float*  hb      = (float*) carve((size_t)Bb * Hh * 4);
    float*  h2f     = (float*) carve((size_t)Bb * Hh * 4);
    float*  h2b     = (float*) carve((size_t)Bb * Hh * 4);

    const int MT = Bb * Tt;                                // 32768 rows
    const int nblk_gemm = (MT / 64) * (G4H / 64) / 8;      // 2048 blocks

    // 1) lengths
    lengths_kernel<<<1, 64, 0, stream>>>(x, d_len);

    // 2) x -> bf16
    {
        const int n = Bb * Tt * Dd;
        f32_to_bf16_kernel<<<(n + 255) / 256, 256, 0, stream>>>(x, xbf, n);
    }

    // ---- layer 1 forward ---------------------------------------------------
    {
        const int n = G4H * Dd;
        f32_to_bf16_kernel<<<(n + 255) / 256, 256, 0, stream>>>(W1f_ih, wbf, n);
        gemm_bf16_wmma_kernel<<<nblk_gemm, 256, 0, stream>>>(xbf, wbf, b1f, xproj, MT, G4H, Dd);
        zero_f32_kernel<<<(Bb * Hh + 255) / 256, 256, 0, stream>>>(hb, Bb * Hh);
        lstm_rec_kernel<<<16, 512, REC_SMEM, stream>>>(xproj, W1f_hh, d_len, hb, out1, 0, 1);
    }
    // ---- layer 1 backward --------------------------------------------------
    {
        const int n = G4H * Dd;
        f32_to_bf16_kernel<<<(n + 255) / 256, 256, 0, stream>>>(W1b_ih, wbf, n);
        gemm_bf16_wmma_kernel<<<nblk_gemm, 256, 0, stream>>>(xbf, wbf, b1b, xproj, MT, G4H, Dd);
        zero_f32_kernel<<<(Bb * Hh + 255) / 256, 256, 0, stream>>>(hb, Bb * Hh);
        lstm_rec_kernel<<<16, 512, REC_SMEM, stream>>>(xproj, W1b_hh, d_len, hb, out1, 1, 2);
    }

    // 3) out1 -> bf16
    {
        const int n = Bb * Tt * Hh;
        f32_to_bf16_kernel<<<(n + 255) / 256, 256, 0, stream>>>(out1, out1bf, n);
    }

    // ---- layer 2 forward (only final h needed) -----------------------------
    {
        const int n = G4H * Hh;
        f32_to_bf16_kernel<<<(n + 255) / 256, 256, 0, stream>>>(W2f_ih, wbf, n);
        gemm_bf16_wmma_kernel<<<nblk_gemm, 256, 0, stream>>>(out1bf, wbf, b2f, xproj, MT, G4H, Hh);
        zero_f32_kernel<<<(Bb * Hh + 255) / 256, 256, 0, stream>>>(h2f, Bb * Hh);
        lstm_rec_kernel<<<16, 512, REC_SMEM, stream>>>(xproj, W2f_hh, d_len, h2f, nullptr, 0, 0);
    }
    // ---- layer 2 backward --------------------------------------------------
    {
        const int n = G4H * Hh;
        f32_to_bf16_kernel<<<(n + 255) / 256, 256, 0, stream>>>(W2b_ih, wbf, n);
        gemm_bf16_wmma_kernel<<<nblk_gemm, 256, 0, stream>>>(out1bf, wbf, b2b, xproj, MT, G4H, Hh);
        zero_f32_kernel<<<(Bb * Hh + 255) / 256, 256, 0, stream>>>(h2b, Bb * Hh);
        lstm_rec_kernel<<<16, 512, REC_SMEM, stream>>>(xproj, W2b_hh, d_len, h2b, nullptr, 1, 0);
    }

    // ---- head --------------------------------------------------------------
    head_kernel<<<(Bb * OUTN + 255) / 256, 256, 0, stream>>>(h2f, h2b, W3, b3, out);
}
